// DeepHandwritingPredictionModel_38096359916088
// MI455X (gfx1250) — matrix operand order repaired and checked by reading, not compiled, not run on hardware
//
#include <hip/hip_runtime.h>
#include <stdint.h>

#define BB 64       // batch
#define TT 800      // time steps
#define UU 400      // hidden units
#define UP 416      // hidden padded to multiple of 32 (K for WMMA)
#define GG 1600     // 4*U gate columns
#define KTI 13      // UP/32 k-tiles
#define NTW 100     // GG/16 n-tiles for gate matrices
#define NWGL 25     // workgroups for the recurrence (one unit-group each)
#define LSTM_THREADS 128
#define YD 121      // MDN output dim (6*M+1)
#define YP 128      // padded
#define NROWS (BB*TT)

typedef __attribute__((ext_vector_type(16))) __bf16 v16bf;
typedef __attribute__((ext_vector_type(8)))  __bf16 v8bf;
typedef __attribute__((ext_vector_type(8)))  float  v8f;

__device__ __forceinline__ float sigf(float v) { return 1.0f / (1.0f + __expf(-v)); }

// ---------------- zero scratch (h buffers, zero-slab, barrier counters) ----------------
__global__ void zero_kernel(uint32_t* __restrict__ p, long n) {
  long i = (long)blockIdx.x * blockDim.x + threadIdx.x;
  long stride = (long)gridDim.x * blockDim.x;
  for (; i < n; i += stride) p[i] = 0u;
}

// ---- pack f32 weight [krows x ncols] (row stride ldn) into WMMA B-fragment order bf16 ----
// Fragment element (kt, nt, lane, e) <- W[k, n] with
//   k = kt*32 + 8*(lane>=16) + (e&7) + 16*(e>>3), n = nt*16 + (lane&15)   (16-bit B 32x16 layout)
__global__ void pack_b_kernel(const float* __restrict__ W, int krows, int ncols, int ldn,
                              int ntiles, __bf16* __restrict__ out, int total) {
  int idx = blockIdx.x * blockDim.x + threadIdx.x;
  if (idx >= total) return;
  int e    = idx & 15;
  int lane = (idx >> 4) & 31;
  int nt   = (idx >> 9) % ntiles;
  int kt   = (idx >> 9) / ntiles;
  int k = kt * 32 + ((lane >> 4) << 3) + (e & 7) + ((e >> 3) << 4);
  int n = nt * 16 + (lane & 15);
  float v = (k < krows && n < ncols) ? W[(size_t)k * ldn + n] : 0.0f;
  out[idx] = (__bf16)v;
}

// ---------------- multi-WGP LSTM layer over all T ----------------
// Grid = 25 workgroups; WG w owns units 16w..16w+15 (its 4 WMMA N-tiles are the
// i/f/c/o gate tiles, identical per-lane (row,unit) mapping -> lane-local gates).
// 4 waves per WG, wave = M-tile. Weights live in LDS; h flows through L2 with one
// device-scope barrier per timestep. A fragments and x rows are prefetched into
// registers so each step exposes only ~one L2 round trip per matmul phase.
__global__ __launch_bounds__(LSTM_THREADS, 1)
void lstm_layer_kernel(const float* __restrict__ x,      // [B,T,3]
                       const float* __restrict__ Wx3,    // first 3 rows of Wx, ld=1600
                       const __bf16* __restrict__ Whp,   // packed recurrent weights
                       const __bf16* __restrict__ Wxp,   // packed h-input weights or null
                       const float* __restrict__ bias,   // [1600]
                       const float* __restrict__ peepI,
                       const float* __restrict__ peepF,
                       const float* __restrict__ peepO,
                       const __bf16* __restrict__ hprev, // [T,64,416] or null
                       const __bf16* __restrict__ hzero, // zeroed [64,416] slab
                       __bf16* __restrict__ hout,        // [T,64,416]
                       unsigned* __restrict__ bar)       // [TT] zeroed counters
{
  // wlds[0] = recurrent Wh fragments, wlds[1] = input-path Wx fragments
  __shared__ __align__(64) __bf16 wlds[2][KTI][4][32][16];  // 106,496 B

  const int tid     = threadIdx.x;
  const int wv      = tid >> 5;     // wave id == M-tile (0..3)
  const int lane    = tid & 31;
  const int lane_n  = lane & 15;
  const int lane_hi = lane >> 4;
  const int wg      = blockIdx.x;   // unit-group (0..24)
  const int u       = wg * 16 + lane_n;
  const bool hasPrev = (Wxp != nullptr);
  const size_t HSTRIDE = (size_t)BB * UP;

  // ---- preload this WG's weight fragments into LDS (once) ----
  {
    uint4* dst0 = (uint4*)&wlds[0][0][0][0][0];
    for (int i = tid; i < (KTI * 4 * 512) / 8; i += LSTM_THREADS) {
      int blk = i >> 6, off = i & 63;            // 64 uint4 per (kt,g) block
      int kt = blk >> 2, g = blk & 3;
      dst0[i] = ((const uint4*)(Whp + ((size_t)(kt * NTW + g * 25 + wg) * 32) * 16))[off];
    }
    if (hasPrev) {
      uint4* dst1 = (uint4*)&wlds[1][0][0][0][0];
      for (int i = tid; i < (KTI * 4 * 512) / 8; i += LSTM_THREADS) {
        int blk = i >> 6, off = i & 63;
        int kt = blk >> 2, g = blk & 3;
        dst1[i] = ((const uint4*)(Wxp + ((size_t)(kt * NTW + g * 25 + wg) * 32) * 16))[off];
      }
    }
  }
  __syncthreads();

  // persistent cell state (this wave's 16 rows x this lane's unit)
  float cst[8];
#pragma unroll
  for (int r = 0; r < 8; ++r) cst[r] = 0.0f;

  const float bi = bias[0 * UU + u];
  const float bf = bias[1 * UU + u];
  const float bc = bias[2 * UU + u];
  const float bo = bias[3 * UU + u];
  const float pi = peepI[u];
  const float pf = peepF[u];
  const float po = peepO[u];
  float wxi[3], wxf[3], wxc[3], wxo[3];
#pragma unroll
  for (int d = 0; d < 3; ++d) {
    wxi[d] = Wx3[d * GG + 0 * UU + u];
    wxf[d] = Wx3[d * GG + 1 * UU + u];
    wxc[d] = Wx3[d * GG + 2 * UU + u];
    wxo[d] = Wx3[d * GG + 3 * UU + u];
  }

  const __bf16* hsrc = hzero;  // h_{-1} = 0

  for (int t = 0; t < TT; ++t) {
    v8f ar[4] = {};  // recurrent-path accumulators (per gate)
    v8f ai[4] = {};  // input-path accumulators (per gate)

    // ---- phase 1: z += h_{t-1} @ Wh ----
    {
      const __bf16* hrow = hsrc + (size_t)(wv * 16 + lane_n) * UP;
      v16bf a[KTI];  // prefetch ALL A fragments (one pipelined L2 round trip)
#pragma unroll
      for (int kt = 0; kt < KTI; ++kt) {
        const int colb = kt * 32 + 8 * lane_hi;
        v8bf lo = *(const v8bf*)(hrow + colb);
        v8bf hi = *(const v8bf*)(hrow + colb + 16);
        a[kt] = __builtin_shufflevector(lo, hi, 0, 1, 2, 3, 4, 5, 6, 7,
                                        8, 9, 10, 11, 12, 13, 14, 15);
      }
#pragma unroll
      for (int kt = 0; kt < KTI; ++kt) {
#pragma unroll
        for (int g = 0; g < 4; ++g) {
          v16bf b = *(const v16bf*)(&wlds[0][kt][g][lane][0]);
          ar[g] = __builtin_amdgcn_wmma_f32_16x16x32_bf16(
              false, a[kt], false, b, (short)0, ar[g], false, false);
        }
      }
    }

    // ---- phase 2: z += hprev_t @ Wx[3:,:]  (layers 1,2) ----
    if (hasPrev) {
      const __bf16* hrow = hprev + (size_t)t * HSTRIDE + (size_t)(wv * 16 + lane_n) * UP;
      v16bf a[KTI];
#pragma unroll
      for (int kt = 0; kt < KTI; ++kt) {
        const int colb = kt * 32 + 8 * lane_hi;
        v8bf lo = *(const v8bf*)(hrow + colb);
        v8bf hi = *(const v8bf*)(hrow + colb + 16);
        a[kt] = __builtin_shufflevector(lo, hi, 0, 1, 2, 3, 4, 5, 6, 7,
                                        8, 9, 10, 11, 12, 13, 14, 15);
      }
#pragma unroll
      for (int kt = 0; kt < KTI; ++kt) {
#pragma unroll
        for (int g = 0; g < 4; ++g) {
          v16bf b = *(const v16bf*)(&wlds[1][kt][g][lane][0]);
          ai[g] = __builtin_amdgcn_wmma_f32_16x16x32_bf16(
              false, a[kt], false, b, (short)0, ai[g], false, false);
        }
      }
    }

    // ---- prefetch x rows for this wave's 8 C-rows (one round trip) ----
    float xv[8][3];
#pragma unroll
    for (int r = 0; r < 8; ++r) {
      const int m = wv * 16 + r + 8 * lane_hi;
      const float* xp = x + ((size_t)m * TT + t) * 3;
      xv[r][0] = xp[0]; xv[r][1] = xp[1]; xv[r][2] = xp[2];
    }

    // ---- lane-local gates, peepholes, state update ----
#pragma unroll
    for (int r = 0; r < 8; ++r) {
      const int m = wv * 16 + r + 8 * lane_hi;  // C-matrix row mapping
      const float x0 = xv[r][0], x1 = xv[r][1], x2 = xv[r][2];
      float zi = ar[0][r] + ai[0][r] + bi + x0 * wxi[0] + x1 * wxi[1] + x2 * wxi[2];
      float zf = ar[1][r] + ai[1][r] + bf + x0 * wxf[0] + x1 * wxf[1] + x2 * wxf[2];
      float zc = ar[2][r] + ai[2][r] + bc + x0 * wxc[0] + x1 * wxc[1] + x2 * wxc[2];
      float zo = ar[3][r] + ai[3][r] + bo + x0 * wxo[0] + x1 * wxo[1] + x2 * wxo[2];
      float c  = cst[r];
      float ig = sigf(zi + pi * c);
      float fg = sigf(zf + pf * c);
      float cn = fg * c + ig * tanhf(zc);
      float og = sigf(zo + po * cn);
      float hv = og * tanhf(cn);
      cst[r] = cn;
      hout[(size_t)t * HSTRIDE + (size_t)m * UP + u] = (__bf16)hv;
    }

    // ---- device-scope barrier across the 25 WGs for step t ----
    __threadfence();      // publish h_t stores to device scope
    __syncthreads();      // all threads of this WG have stored + fenced
    if (tid == 0) {
      __hip_atomic_fetch_add(&bar[t], 1u, __ATOMIC_ACQ_REL, __HIP_MEMORY_SCOPE_AGENT);
      while (__hip_atomic_load(&bar[t], __ATOMIC_ACQUIRE, __HIP_MEMORY_SCOPE_AGENT) < NWGL) {
        __builtin_amdgcn_s_sleep(1);
      }
    }
    __syncthreads();
    __threadfence();      // acquire: h_t from all WGs now readable

    hsrc = hout + (size_t)t * HSTRIDE;  // becomes h_{t-1} for next step
  }
}

// ---------------- MDN head GEMM: [51200,416]bf16 @ packed[416,128] -> y f32 ----------------
__global__ __launch_bounds__(256)
void mdn_gemm_kernel(const __bf16* __restrict__ h,   // [NROWS][UP]
                     const __bf16* __restrict__ Wp,  // packed [13][8][32][16]
                     const float* __restrict__ bmdn, // [121]
                     float* __restrict__ y)          // [NROWS][YP]
{
  const int wave = threadIdx.x >> 5, lane = threadIdx.x & 31;
  const int lane_n = lane & 15, lane_hi = lane >> 4;
  const int mtile = blockIdx.x * 8 + wave;  // 0..3199
  const __bf16* hrow = h + (size_t)(mtile * 16 + lane_n) * UP;
  v8f acc[8] = {};
#pragma unroll 1
  for (int kt = 0; kt < KTI; ++kt) {
    const int colb = kt * 32 + 8 * lane_hi;
    v8bf lo = *(const v8bf*)(hrow + colb);
    v8bf hi = *(const v8bf*)(hrow + colb + 16);
    v16bf a = __builtin_shufflevector(lo, hi, 0, 1, 2, 3, 4, 5, 6, 7,
                                      8, 9, 10, 11, 12, 13, 14, 15);
#pragma unroll
    for (int nt = 0; nt < 8; ++nt) {
      v16bf b = *(const v16bf*)(Wp + ((size_t)(kt * 8 + nt) * 32 + lane) * 16);
      acc[nt] = __builtin_amdgcn_wmma_f32_16x16x32_bf16(
          false, a, false, b, (short)0, acc[nt], false, false);
    }
  }
#pragma unroll
  for (int nt = 0; nt < 8; ++nt) {
    const int col = nt * 16 + lane_n;
    const float bv = (col < YD) ? bmdn[col] : 0.0f;
#pragma unroll
    for (int r = 0; r < 8; ++r) {
      const int row = mtile * 16 + r + 8 * lane_hi;
      y[(size_t)row * YP + col] = acc[nt][r] + bv;
    }
  }
}

// ---------------- MDN transforms: softmax / exp / tanh / sigmoid, permute to [B,T,121] ----
__global__ void mdn_transform_kernel(const float* __restrict__ y, float* __restrict__ out) {
  int row = blockIdx.x * blockDim.x + threadIdx.x;  // row = t*64 + m
  if (row >= NROWS) return;
  int t = row >> 6, m = row & 63;
  const float* yr = y + (size_t)row * YP;
  float* o = out + ((size_t)m * TT + t) * YD;
  float mx = -3.402823e38f;
  float e[20];
#pragma unroll
  for (int j = 0; j < 20; ++j) mx = fmaxf(mx, yr[j]);
  float s = 0.0f;
#pragma unroll
  for (int j = 0; j < 20; ++j) { e[j] = __expf(yr[j] - mx); s += e[j]; }
  float inv = 1.0f / s;
#pragma unroll
  for (int j = 0; j < 20; ++j) o[j] = e[j] * inv;          // pi
  for (int j = 20; j < 60; ++j) o[j] = yr[j];              // mu1, mu2
  for (int j = 60; j < 100; ++j) o[j] = __expf(yr[j]);     // s1, s2
  for (int j = 100; j < 120; ++j) o[j] = tanhf(yr[j]);     // rho
  o[120] = sigf(yr[120]);                                  // eos
}

extern "C" void kernel_launch(void* const* d_in, const int* in_sizes, int n_in,
                              void* d_out, int out_size, void* d_ws, size_t ws_size,
                              hipStream_t stream) {
  (void)in_sizes; (void)n_in; (void)out_size; (void)ws_size;
  const float* x    = (const float*)d_in[0];
  const float* Wx0  = (const float*)d_in[1];
  const float* Wh0  = (const float*)d_in[2];
  const float* b0   = (const float*)d_in[3];
  const float* pi0  = (const float*)d_in[4];
  const float* pf0  = (const float*)d_in[5];
  const float* po0  = (const float*)d_in[6];
  const float* Wx1  = (const float*)d_in[7];
  const float* Wh1  = (const float*)d_in[8];
  const float* b1   = (const float*)d_in[9];
  const float* pi1  = (const float*)d_in[10];
  const float* pf1  = (const float*)d_in[11];
  const float* po1  = (const float*)d_in[12];
  const float* Wx2  = (const float*)d_in[13];
  const float* Wh2  = (const float*)d_in[14];
  const float* b2   = (const float*)d_in[15];
  const float* pi2  = (const float*)d_in[16];
  const float* pf2  = (const float*)d_in[17];
  const float* po2  = (const float*)d_in[18];
  const float* Wmdn = (const float*)d_in[19];
  const float* bmdn = (const float*)d_in[20];

  // ---- carve workspace ----
  char* ws = (char*)d_ws;
  size_t off = 0;
  auto carve = [&](size_t bytes) -> char* {
    char* p = ws + off;
    off += (bytes + 255) & ~(size_t)255;
    return p;
  };
  const size_t HB  = (size_t)TT * BB * UP * 2;   // 42,598,400 B (multiple of 256)
  const size_t HZ  = (size_t)BB * UP * 2;        // 53,248 B (multiple of 256)
  const size_t BARB = (size_t)3 * TT * 4;        // 9,600 B barrier counters
  __bf16*   hbuf0 = (__bf16*)carve(HB);
  __bf16*   hbuf1 = (__bf16*)carve(HB);          // contiguous with hbuf0
  __bf16*   hzero = (__bf16*)carve(HZ);          // contiguous
  unsigned* barb  = (unsigned*)carve(BARB);      // contiguous
  const size_t ZEROB = 2 * HB + HZ + ((BARB + 255) & ~(size_t)255);
  const size_t WHPB = (size_t)KTI * NTW * 32 * 16 * 2;
  __bf16* Whp0 = (__bf16*)carve(WHPB);
  __bf16* Whp1 = (__bf16*)carve(WHPB);
  __bf16* Whp2 = (__bf16*)carve(WHPB);
  __bf16* Wxp1 = (__bf16*)carve(WHPB);
  __bf16* Wxp2 = (__bf16*)carve(WHPB);
  const size_t WMPB = (size_t)KTI * 8 * 32 * 16 * 2;
  __bf16* Wmdnp = (__bf16*)carve(WMPB);
  float* ybuf = (float*)carve((size_t)NROWS * YP * 4);

  // zero h buffers (K-pad columns), zero-slab, and barrier counters
  zero_kernel<<<4096, 256, 0, stream>>>((uint32_t*)hbuf0, (long)(ZEROB / 4));

  // pack weights into B-fragment order (bf16)
  const int totalH = KTI * NTW * 512;
  const int blkH = (totalH + 255) / 256;
  pack_b_kernel<<<blkH, 256, 0, stream>>>(Wh0, UU, GG, GG, NTW, Whp0, totalH);
  pack_b_kernel<<<blkH, 256, 0, stream>>>(Wh1, UU, GG, GG, NTW, Whp1, totalH);
  pack_b_kernel<<<blkH, 256, 0, stream>>>(Wh2, UU, GG, GG, NTW, Whp2, totalH);
  pack_b_kernel<<<blkH, 256, 0, stream>>>(Wx1 + 3 * GG, UU, GG, GG, NTW, Wxp1, totalH);
  pack_b_kernel<<<blkH, 256, 0, stream>>>(Wx2 + 3 * GG, UU, GG, GG, NTW, Wxp2, totalH);
  const int totalM = KTI * 8 * 512;
  pack_b_kernel<<<(totalM + 255) / 256, 256, 0, stream>>>(Wmdn, UU, YD, YD, 8, Wmdnp, totalM);

  // sequential recurrent layers: 25 WGPs cooperating per step via device barrier
  lstm_layer_kernel<<<NWGL, LSTM_THREADS, 0, stream>>>(
      x, Wx0, Whp0, nullptr, b0, pi0, pf0, po0, nullptr, hzero, hbuf0, barb + 0 * TT);
  lstm_layer_kernel<<<NWGL, LSTM_THREADS, 0, stream>>>(
      x, Wx1, Whp1, Wxp1, b1, pi1, pf1, po1, hbuf0, hzero, hbuf1, barb + 1 * TT);
  lstm_layer_kernel<<<NWGL, LSTM_THREADS, 0, stream>>>(
      x, Wx2, Whp2, Wxp2, b2, pi2, pf2, po2, hbuf1, hzero, hbuf0, barb + 2 * TT);

  // MDN head (parallel across all WGPs)
  mdn_gemm_kernel<<<NROWS / 16 / 8, 256, 0, stream>>>(hbuf0, Wmdnp, bmdn, ybuf);
  mdn_transform_kernel<<<(NROWS + 255) / 256, 256, 0, stream>>>(ybuf, (float*)d_out);
}
